// SelectiveScanIC_84567906058584
// MI455X (gfx1250) — matrix operand order repaired and checked by reading, not compiled, not run on hardware
//
#include <hip/hip_runtime.h>

// Problem dims (fixed by the reference)
#define NB 4
#define NK 4096
#define ND 2048
#define TILE_E 64
#define TILE_M 32   // two 16-row WMMA M-tiles per chunk -> B operand reused 2x
#define WPAD 2056   // 2048 + 8 f16 row pad -> +4 banks/row, conflict-free b128 reads
#define XQ   256    // x-chunk staged in K-segments of 256
#define XQPAD 264   // 256 + 8 f16 row pad

typedef __attribute__((ext_vector_type(16))) _Float16 v16h;
typedef __attribute__((ext_vector_type(8)))  _Float16 v8h;
typedef __attribute__((ext_vector_type(4)))  _Float16 v4h;
typedef __attribute__((ext_vector_type(8)))  float    v8f;

union ABOp { v16h v; v8h h[2]; };

__global__ __launch_bounds__(128, 1)
void fused_gemm_sigmoid_scan(const float* __restrict__ xg,
                             const float* __restrict__ Wg,
                             const float* __restrict__ bg,
                             float* __restrict__ outg) {
  // 257KB W tile + 2 x 16.5KB double-buffered x tile = ~290KB of the 320KB WGP LDS
  __shared__ _Float16 wLDS[TILE_E * WPAD];
  __shared__ _Float16 xLDS[2][TILE_M * XQPAD];

  const int tid  = threadIdx.x;
  const int wave = tid >> 5;
  const int lane = tid & 31;
  const int ln   = lane & 15;
  const bool hi  = lane >= 16;

  const int batch = blockIdx.x >> 5;          // 32 channel tiles per batch
  const int e0    = (blockIdx.x & 31) * TILE_E;

  // ---- stage W tile (64 x 2048) f32 -> f16 into LDS (one time) ----
  for (int idx = tid; idx < TILE_E * (ND / 4); idx += 128) {
    const int row = idx >> 9;                 // ND/4 = 512 float4 per row
    const int c4  = idx & 511;
    const float4 w4 =
        reinterpret_cast<const float4*>(Wg + (size_t)(e0 + row) * ND)[c4];
    v4h h;
    h.x = (_Float16)w4.x; h.y = (_Float16)w4.y;
    h.z = (_Float16)w4.z; h.w = (_Float16)w4.w;
    *reinterpret_cast<v4h*>(&wLDS[row * WPAD + c4 * 4]) = h;
  }

  const int   e    = e0 + wave * 16 + ln;     // this lane's output channel
  const float bias = bg[e];

  // ISA operand layouts (wave32):
  // A 16x32 f16: lanes0-15 K={0..7,16..23}, lanes16-31 K={8..15,24..31}
  // B 32x16 f16: lanes0-15 K={0..7,16..23}, lanes16-31 K={8..15,24..31}
  const int koffA = hi ? 8 : 0;
  const int koffB = hi ? 8 : 0;
  const _Float16* bRow = &wLDS[(wave * 16 + ln) * WPAD];

  // --- register-staged prefetch of one x segment (32 rows x 256 cols f32) ---
  const int srow = tid >> 2;                  // 4 threads per row
  const int scol = tid & 3;                   // 16 float4 per thread, stride 4
  float4 R[16];

  auto loadQ = [&](int k0q, int q) {
#pragma unroll
    for (int j = 0; j < 16; ++j) {
      R[j] = reinterpret_cast<const float4*>(
          xg + (size_t)(batch * NK + k0q + srow) * ND + q * XQ)[scol + j * 4];
    }
  };
  auto storeQ = [&](int buf) {
    _Float16* dst = &xLDS[buf][srow * XQPAD];
#pragma unroll
    for (int j = 0; j < 16; ++j) {
      v4h h;
      h.x = (_Float16)R[j].x; h.y = (_Float16)R[j].y;
      h.z = (_Float16)R[j].z; h.w = (_Float16)R[j].w;
      *reinterpret_cast<v4h*>(dst + (scol + j * 4) * 4) = h;
    }
  };

  float s = 0.0f;  // recurrence state for channel e (replicated in both lane halves)

  loadQ(0, 0);     // prologue prefetch

  for (int k0 = 0; k0 < NK; k0 += TILE_M) {
    // 2 M-tiles x 2 chains: same-chain WMMA distance = 4 (covers the RAW hazard)
    v8f acc0[2] = {}, acc1[2] = {};

    // prefetch u_t early: latency hides under the 128 WMMAs below
    const int    mBase = hi ? 8 : 0;
    const size_t gbase = (size_t)(batch * NK + k0 + mBase) * ND + e;
    float u0[8], u1[8];
#pragma unroll
    for (int r = 0; r < 8; ++r) {
      u0[r] = xg[gbase + (size_t)r * ND];
      u1[r] = xg[gbase + (size_t)(r + 16) * ND];
    }

#pragma unroll
    for (int q = 0; q < ND / XQ; ++q) {       // 8 K-segments of 256
      const int buf = q & 1;
      storeQ(buf);                            // publish prefetched segment
      if (q < ND / XQ - 1)        loadQ(k0, q + 1);       // next segment
      else if (k0 + TILE_M < NK)  loadQ(k0 + TILE_M, 0);  // next chunk
      __syncthreads();

      const _Float16* aRow0 = &xLDS[buf][ln * XQPAD];
      const _Float16* aRow1 = aRow0 + 16 * XQPAD;
      const _Float16* bQ    = bRow + q * XQ;
#pragma unroll
      for (int i = 0; i < XQ / 32; ++i) {     // 8 WMMA steps, 2 WMMAs each
        const int kk = i * 32;
        ABOp a0, a1, bm;
        bm.h[0] = *reinterpret_cast<const v8h*>(bQ + kk + koffB);
        bm.h[1] = *reinterpret_cast<const v8h*>(bQ + kk + 16 + koffB);
        a0.h[0] = *reinterpret_cast<const v8h*>(aRow0 + kk + koffA);
        a0.h[1] = *reinterpret_cast<const v8h*>(aRow0 + kk + 16 + koffA);
        a1.h[0] = *reinterpret_cast<const v8h*>(aRow1 + kk + koffA);
        a1.h[1] = *reinterpret_cast<const v8h*>(aRow1 + kk + 16 + koffA);
        acc0[i & 1] = __builtin_amdgcn_wmma_f32_16x16x32_f16(
            false, a0.v, false, bm.v, (short)0, acc0[i & 1], false, false);
        acc1[i & 1] = __builtin_amdgcn_wmma_f32_16x16x32_f16(
            false, a1.v, false, bm.v, (short)0, acc1[i & 1], false, false);
      }
    }

    // merge partial-K accumulators: lam pre-activations
    // C layout: lanes0-15 -> m = r (n = lane), lanes16-31 -> m = r + 8
    v8f c0 = acc0[0] + acc0[1];               // timesteps k0 + 0..15
    v8f c1 = acc1[0] + acc1[1];               // timesteps k0 + 16..31

    float lam0[8], lam1[8];
#pragma unroll
    for (int r = 0; r < 8; ++r) {
      lam0[r] = 1.0f / (1.0f + __expf(-(c0[r] + bias)));
      lam1[r] = 1.0f / (1.0f + __expf(-(c1[r] + bias)));
    }

    float outv0[8], outv1[8];
    // phase 0: m = 0..7 (valid in low lanes)
    float sA = s;
#pragma unroll
    for (int r = 0; r < 8; ++r) {
      sA = lam0[r] * sA + (1.0f - lam0[r]) * u0[r];
      outv0[r] = sA;
    }
    float sv = __int_as_float(
        __builtin_amdgcn_ds_bpermute(ln * 4, __float_as_int(sA)));
    // phase 1: m = 8..15 (valid in high lanes)
#pragma unroll
    for (int r = 0; r < 8; ++r) {
      sv = lam0[r] * sv + (1.0f - lam0[r]) * u0[r];
      if (hi) outv0[r] = sv;
    }
    sv = __int_as_float(
        __builtin_amdgcn_ds_bpermute((ln + 16) * 4, __float_as_int(sv)));
    // phase 2: m = 16..23 (valid in low lanes)
#pragma unroll
    for (int r = 0; r < 8; ++r) {
      sv = lam1[r] * sv + (1.0f - lam1[r]) * u1[r];
      outv1[r] = sv;
    }
    float sv2 = __int_as_float(
        __builtin_amdgcn_ds_bpermute(ln * 4, __float_as_int(sv)));
    // phase 3: m = 24..31 (valid in high lanes)
#pragma unroll
    for (int r = 0; r < 8; ++r) {
      sv2 = lam1[r] * sv2 + (1.0f - lam1[r]) * u1[r];
      if (hi) outv1[r] = sv2;
    }
    s = __int_as_float(
        __builtin_amdgcn_ds_bpermute((ln + 16) * 4, __float_as_int(sv2)));

    // non-temporal: don't let the 128MB output stream evict L2-resident x
#pragma unroll
    for (int r = 0; r < 8; ++r) {
      __builtin_nontemporal_store(outv0[r], &outg[gbase + (size_t)r * ND]);
      __builtin_nontemporal_store(outv1[r], &outg[gbase + (size_t)(r + 16) * ND]);
    }
  }
}

extern "C" void kernel_launch(void* const* d_in, const int* in_sizes, int n_in,
                              void* d_out, int out_size, void* d_ws, size_t ws_size,
                              hipStream_t stream) {
  (void)in_sizes; (void)n_in; (void)out_size; (void)d_ws; (void)ws_size;
  const float* x = (const float*)d_in[0];
  const float* W = (const float*)d_in[1];
  const float* b = (const float*)d_in[2];
  float* out = (float*)d_out;
  // 128 WGs = 4 batches x 32 channel tiles; 128 threads = 4 wave32
  fused_gemm_sigmoid_scan<<<dim3(128), dim3(128), 0, stream>>>(x, W, b, out);
}
